// RGCN_34935263986356
// MI455X (gfx1250) — compile-verified, matrix operand order checked by reading, not run on hardware
//
#include <hip/hip_runtime.h>
#include <hip/hip_bf16.h>

#define N_RELS 8

typedef __attribute__((ext_vector_type(16))) __bf16 v16bf;
typedef __attribute__((ext_vector_type(8)))  float  v8f;

// ---------------- edge-count per (dst, rel) ----------------
__global__ void __launch_bounds__(256)
rgcn_count(const int* __restrict__ dst, const int* __restrict__ et,
           float* __restrict__ cnt, int nE) {
  int e = blockIdx.x * blockDim.x + threadIdx.x;
  if (e < nE) atomicAdd(cnt + (size_t)dst[e] * N_RELS + et[e], 1.0f);
}

// cnt -> 1/max(cnt,1) in place
__global__ void __launch_bounds__(256)
rgcn_inv(float* __restrict__ cnt, int n) {
  int i = blockIdx.x * blockDim.x + threadIdx.x;
  if (i < n) cnt[i] = 1.0f / fmaxf(cnt[i], 1.0f);
}

// ---------------- weight swizzle: f32 [m][K][ncols] -> bf16 B fragments ----
// Fragment layout: frag = (m*2+kh)*ctn + ct, 512 bf16 per frag = [lane][16].
// Lane l holds column n = ct*16 + (l&15); element j holds
// k = kh*32 + ((l&16)?8:0) + (j<8 ? j : 16+(j-8))  (ISA 16-bit B layout).
__global__ void __launch_bounds__(256)
rgcn_swizzle(const float* __restrict__ src, int nmat, int ncols, int ctn,
             __bf16* __restrict__ dstp) {
  int idx = blockIdx.x * blockDim.x + threadIdx.x;
  int total = nmat * 2 * ctn * 512;
  if (idx >= total) return;
  int j    = idx & 15;
  int lane = (idx >> 4) & 31;
  int frag = idx >> 9;
  int ct = frag % ctn;
  int kh = (frag / ctn) & 1;
  int m  = frag / (ctn * 2);
  int n  = ct * 16 + (lane & 15);
  int kb = (lane & 16) ? 8 : 0;
  int k  = kh * 32 + kb + ((j & 8) ? ((j & 7) + 16) : j);
  dstp[idx] = (__bf16)src[((size_t)m * 64 + k) * ncols + n];
}

// ---------------- per-edge scatter-add of 64-float source row -------------
// One wave per edge; each lane adds 2 floats.
__global__ void __launch_bounds__(256)
rgcn_scatter(const float* __restrict__ feat, const int* __restrict__ srcI,
             const int* __restrict__ dstI, const int* __restrict__ et,
             float* __restrict__ S, int nE) {
  int wave = (blockIdx.x * blockDim.x + threadIdx.x) >> 5;
  int lane = threadIdx.x & 31;
  if (wave >= nE) return;
  int s = srcI[wave], d = dstI[wave], r = et[wave];
  const float2 v = ((const float2*)(feat + (size_t)s * 64))[lane];
  float* base = S + ((size_t)d * N_RELS + r) * 64 + lane * 2;
  atomicAdd(base, v.x);
  atomicAdd(base + 1, v.y);
}

// ---------------- A-fragment loader (ISA 16-bit A layout) -----------------
__device__ __forceinline__ v16bf load_a_frag(const float* __restrict__ rowp,
                                             int base, float sc) {
  alignas(16) float t[16];
  *(float4*)(t + 0)  = *(const float4*)(rowp + base);
  *(float4*)(t + 4)  = *(const float4*)(rowp + base + 4);
  *(float4*)(t + 8)  = *(const float4*)(rowp + base + 16);
  *(float4*)(t + 12) = *(const float4*)(rowp + base + 20);
  v16bf a;
#pragma unroll
  for (int j = 0; j < 16; ++j) a[j] = (__bf16)(t[j] * sc);
  return a;
}

// ---------------- fused mean + Sum_r(W_r) + root + bias (+ReLU) via WMMA --
template <int CT, bool RELU>
__global__ void __launch_bounds__(128)
rgcn_transform(const float* __restrict__ S, const float* __restrict__ inv,
               const float* __restrict__ xin, const __bf16* __restrict__ swz,
               const float* __restrict__ bias, float* __restrict__ outp,
               int nTiles) {
  extern __shared__ __bf16 smem[];
  const int NFRAG = (N_RELS + 1) * 2 * CT;
  {
    // stage all B fragments for this layer into LDS (16B vector copies)
    const uint4* g = (const uint4*)swz;
    uint4* sm = (uint4*)smem;
    const int nvec = NFRAG * 64;  // NFRAG*512*2B / 16B
    for (int i = threadIdx.x; i < nvec; i += blockDim.x) sm[i] = g[i];
  }
  __syncthreads();

  int wave = threadIdx.x >> 5;
  int lane = threadIdx.x & 31;
  int tile = blockIdx.x * (blockDim.x >> 5) + wave;
  if (tile >= nTiles) return;
  int n0  = tile * 16;
  int row = n0 + (lane & 15);
  int kb  = (lane & 16) ? 8 : 0;

  v8f acc[CT];
#pragma unroll
  for (int c = 0; c < CT; ++c)
#pragma unroll
    for (int j = 0; j < 8; ++j) acc[c][j] = 0.0f;

  // 8 relation matrices: mean_r = S[row,r,:] * inv(cnt)
#pragma unroll
  for (int m = 0; m < N_RELS; ++m) {
    float sc = inv[(size_t)row * N_RELS + m];
    const float* rowp = S + ((size_t)row * N_RELS + m) * 64;
#pragma unroll
    for (int kh = 0; kh < 2; ++kh) {
      v16bf a = load_a_frag(rowp, kh * 32 + kb, sc);
#pragma unroll
      for (int c = 0; c < CT; ++c) {
        v16bf b = *(const v16bf*)(smem + (((m * 2 + kh) * CT + c) << 9) + lane * 16);
        acc[c] = __builtin_amdgcn_wmma_f32_16x16x32_bf16(
            false, a, false, b, (short)0, acc[c], false, false);
      }
    }
  }
  // root transform on the layer input
  {
    const float* rowp = xin + (size_t)row * 64;
#pragma unroll
    for (int kh = 0; kh < 2; ++kh) {
      v16bf a = load_a_frag(rowp, kh * 32 + kb, 1.0f);
#pragma unroll
      for (int c = 0; c < CT; ++c) {
        v16bf b = *(const v16bf*)(smem + (((N_RELS * 2 + kh) * CT + c) << 9) + lane * 16);
        acc[c] = __builtin_amdgcn_wmma_f32_16x16x32_bf16(
            false, a, false, b, (short)0, acc[c], false, false);
      }
    }
  }

  // epilogue: C/D layout -> row = n0 + ((lane>=16)?8:0)+i, col = c*16+(lane&15)
  const int W = CT * 16;
  int rbase = n0 + kb;
#pragma unroll
  for (int c = 0; c < CT; ++c) {
    int col = c * 16 + (lane & 15);
    float bb = bias[col];
#pragma unroll
    for (int i = 0; i < 8; ++i) {
      float v = acc[c][i] + bb;
      if (RELU) v = fmaxf(v, 0.0f);
      outp[(size_t)(rbase + i) * W + col] = v;
    }
  }
}

extern "C" void kernel_launch(void* const* d_in, const int* in_sizes, int n_in,
                              void* d_out, int out_size, void* d_ws, size_t ws_size,
                              hipStream_t stream) {
  const float* x  = (const float*)d_in[0];
  const int*   ei = (const int*)d_in[1];
  const int*   et = (const int*)d_in[2];
  const float* w1 = (const float*)d_in[3];
  const float* r1 = (const float*)d_in[4];
  const float* b1 = (const float*)d_in[5];
  const float* w2 = (const float*)d_in[6];
  const float* r2 = (const float*)d_in[7];
  const float* b2 = (const float*)d_in[8];
  float* outp = (float*)d_out;

  const int nN = in_sizes[0] / 64;
  const int nE = in_sizes[2];
  const int* src = ei;
  const int* dst = ei + nE;

  char* ws = (char*)d_ws;
  size_t off = 0;
  float* Sbuf = (float*)(ws + off); off += (size_t)nN * N_RELS * 64 * sizeof(float);
  float* cnt  = (float*)(ws + off); off += (size_t)nN * N_RELS * sizeof(float);
  float* H    = (float*)(ws + off); off += (size_t)nN * 64 * sizeof(float);
  __bf16* swz1 = (__bf16*)(ws + off); off += (size_t)(N_RELS + 1) * 2 * 4 * 512 * sizeof(__bf16);
  __bf16* swz2 = (__bf16*)(ws + off); off += (size_t)(N_RELS + 1) * 2 * 2 * 512 * sizeof(__bf16);
  (void)ws_size; (void)n_in; (void)out_size;

  hipMemsetAsync(Sbuf, 0, (size_t)nN * N_RELS * 64 * sizeof(float), stream);
  hipMemsetAsync(cnt,  0, (size_t)nN * N_RELS * sizeof(float), stream);

  // swizzle weights into bf16 WMMA B-fragments (once per call; tiny)
  { int t = 8 * 2 * 4 * 512; rgcn_swizzle<<<(t + 255) / 256, 256, 0, stream>>>(w1, 8, 64, 4, swz1); }
  { int t = 1 * 2 * 4 * 512; rgcn_swizzle<<<(t + 255) / 256, 256, 0, stream>>>(r1, 1, 64, 4, swz1 + 8 * 2 * 4 * 512); }
  { int t = 8 * 2 * 2 * 512; rgcn_swizzle<<<(t + 255) / 256, 256, 0, stream>>>(w2, 8, 32, 2, swz2); }
  { int t = 1 * 2 * 2 * 512; rgcn_swizzle<<<(t + 255) / 256, 256, 0, stream>>>(r2, 1, 32, 2, swz2 + 8 * 2 * 2 * 512); }

  rgcn_count<<<(nE + 255) / 256, 256, 0, stream>>>(dst, et, cnt, nE);
  rgcn_inv<<<(nN * N_RELS + 255) / 256, 256, 0, stream>>>(cnt, nN * N_RELS);

  const int nTiles = nN / 16;
  const int scatBlocks = (nE * 32 + 255) / 256;
  const int xfBlocks = (nTiles + 3) / 4;
  const size_t sh1 = (size_t)(N_RELS + 1) * 2 * 4 * 512 * sizeof(__bf16);  // 72 KB
  const size_t sh2 = (size_t)(N_RELS + 1) * 2 * 2 * 512 * sizeof(__bf16);  // 36 KB

  // layer 1
  rgcn_scatter<<<scatBlocks, 256, 0, stream>>>(x, src, dst, et, Sbuf, nE);
  rgcn_transform<4, true><<<xfBlocks, 128, sh1, stream>>>(Sbuf, cnt, x, swz1, b1, H, nTiles);

  // layer 2
  hipMemsetAsync(Sbuf, 0, (size_t)nN * N_RELS * 64 * sizeof(float), stream);
  rgcn_scatter<<<scatBlocks, 256, 0, stream>>>(H, src, dst, et, Sbuf, nE);
  rgcn_transform<2, false><<<xfBlocks, 128, sh2, stream>>>(Sbuf, cnt, H, swz2, b2, outp, nTiles);
}
